// EmbeddingBlock_10883447128784
// MI455X (gfx1250) — matrix operand order repaired and checked by reading, not compiled, not run on hardware
//
#include <hip/hip_runtime.h>

#define N_NODES 16384
#define N_EDGES 262144
#define NUM_RADIAL 6
#define HID 256
#define CND 64
#define HC (HID + CND)            // 320
#define KTOT (3 * HID + 2 * CND)  // 896
#define MT 32                     // edges per block
#define LDSK (KTOT + 8)           // padded LDS row stride (bf16 elems) -> 1808B, 16B aligned

#define AS1 __attribute__((address_space(1)))
#define AS3 __attribute__((address_space(3)))

#if __has_builtin(__builtin_amdgcn_global_load_async_to_lds_b128)
#define USE_ASYNC_STAGE 1
#else
#define USE_ASYNC_STAGE 0
#endif

typedef __attribute__((ext_vector_type(16))) __bf16 v16bf;
typedef __attribute__((ext_vector_type(8)))  __bf16 v8bf;
typedef __attribute__((ext_vector_type(8)))  float  v8f;
typedef __attribute__((ext_vector_type(4)))  int    v4i;

__device__ __forceinline__ unsigned short f2bfu(float f) {
    unsigned u = __float_as_uint(f);
    u += 0x7FFFu + ((u >> 16) & 1u);   // round-to-nearest-even
    return (unsigned short)(u >> 16);
}

__device__ __forceinline__ float fast_swish(float v) {
    return __fdividef(v, 1.f + __expf(-v));
}

__device__ __forceinline__ void wait_asynccnt0() {
#if __has_builtin(__builtin_amdgcn_s_wait_asynccnt)
    __builtin_amdgcn_s_wait_asynccnt(0);
#else
    asm volatile("s_wait_asynccnt 0x0" ::: "memory");
#endif
}

// ---- prep: h[n] = concat(bf16(emb_w[x[n]]), bf16(cond[n]))  -> [N_NODES][320] bf16
__global__ void build_h_kernel(const int* __restrict__ x,
                               const float* __restrict__ cond,
                               const float* __restrict__ emb_w,
                               unsigned short* __restrict__ h) {
    const int n = blockIdx.x;
    const int c = threadIdx.x;                       // 0..319
    float v = (c < HID) ? emb_w[x[n] * HID + c]
                        : cond[n * CND + (c - HID)];
    h[n * HC + c] = f2bfu(v);
}

// ---- prep: w_lin f32[256][896] -> bf16 row-major (this IS the WMMA B layout feed)
__global__ void build_wlin_kernel(const float* __restrict__ w,
                                  unsigned short* __restrict__ wb) {
    const int idx = blockIdx.x * 256 + threadIdx.x;
    wb[idx] = f2bfu(w[idx]);
}

// ---- fused edge MLP: async gather -> LDS feat tile -> bf16 WMMA GEMM -> swish -> NT store
__global__ __launch_bounds__(256)
void edge_mlp_kernel(const unsigned short* __restrict__ h,      // [N][320] bf16
                     const unsigned short* __restrict__ wlinb,  // [256][896] bf16
                     const int* __restrict__ ei,
                     const int* __restrict__ ej,
                     const float* __restrict__ rbf,
                     const float* __restrict__ w_rbf,           // [256][6]
                     const float* __restrict__ b_rbf,           // [256]
                     const float* __restrict__ b_lin,           // [256]
                     float* __restrict__ out) {                 // [E][256]
    __shared__ unsigned short sA[MT * LDSK];        // feat tile, bf16 bits
    __shared__ float sWr[HID * NUM_RADIAL];
    __shared__ float sBr[HID];

    const int tid   = threadIdx.x;
    const int eBase = blockIdx.x * MT;

    // preload small rbf-projection weights
    for (int t = tid; t < HID * NUM_RADIAL; t += 256) sWr[t] = w_rbf[t];
    if (tid < HID) sBr[tid] = b_rbf[tid];

    // stage gathered h rows into feat cols [0, 640): 8 threads/edge, 16B chunks
    {
        const int el  = tid >> 3;       // local edge 0..31
        const int sub = tid & 7;
        const int e   = eBase + el;
        const uint4* si = (const uint4*)(h + (size_t)ei[e] * HC);
        const uint4* sj = (const uint4*)(h + (size_t)ej[e] * HC);
        uint4* dst = (uint4*)(sA + el * LDSK);
#if USE_ASYNC_STAGE
        // ASYNCcnt-tracked DMA into LDS: no VGPR round-trip; overlaps with r-phase
        #pragma unroll
        for (int c = 0; c < 5; ++c)
            __builtin_amdgcn_global_load_async_to_lds_b128(
                (AS1 v4i*)(si + sub * 5 + c),
                (AS3 v4i*)(dst + sub * 5 + c), 0, 0);
        #pragma unroll
        for (int c = 0; c < 5; ++c)
            __builtin_amdgcn_global_load_async_to_lds_b128(
                (AS1 v4i*)(sj + sub * 5 + c),
                (AS3 v4i*)(dst + 40 + sub * 5 + c), 0, 0);
#else
        #pragma unroll
        for (int c = 0; c < 5; ++c) dst[sub * 5 + c]      = si[sub * 5 + c];
        #pragma unroll
        for (int c = 0; c < 5; ++c) dst[40 + sub * 5 + c] = sj[sub * 5 + c];
#endif
    }
    __syncthreads();   // sWr/sBr visible; async gathers may still be in flight

    // r[e] = swish(rbf @ w_rbf^T + b_rbf) -> feat cols [640, 896)  (disjoint LDS region)
    {
        const int el = tid >> 3;
        const int n0 = (tid & 7) * 32;
        const int e  = eBase + el;
        float rv[NUM_RADIAL];
        #pragma unroll
        for (int q = 0; q < NUM_RADIAL; ++q) rv[q] = rbf[e * NUM_RADIAL + q];
        unsigned short* dstr = sA + el * LDSK + 2 * HID + 2 * CND;  // elem 640
        #pragma unroll 4
        for (int n = n0; n < n0 + 32; n += 2) {
            float a0 = sBr[n], a1 = sBr[n + 1];
            #pragma unroll
            for (int q = 0; q < NUM_RADIAL; ++q) {
                a0 += rv[q] * sWr[n * NUM_RADIAL + q];
                a1 += rv[q] * sWr[(n + 1) * NUM_RADIAL + q];
            }
            a0 = fast_swish(a0);
            a1 = fast_swish(a1);
            unsigned pk = (unsigned)f2bfu(a0) | ((unsigned)f2bfu(a1) << 16);
            *(unsigned*)(dstr + n) = pk;
        }
    }
#if USE_ASYNC_STAGE
    wait_asynccnt0();   // this wave's async LDS writes have landed
#endif
    __syncthreads();

    // ---- WMMA GEMM: each wave = 32 edges (2 M-subtiles) x 32 cols (2 N-tiles)
    //      -> every A and B fragment feeds 2 WMMAs; no intra-block B duplication.
    const int lane = tid & 31;
    const int ngrp = tid >> 5;               // wave id = N group (8 x 32 cols)
    const int l15  = lane & 15;
    const int hA   = (lane < 16) ? 0 : 16;   // A-frag: lane halves take K+0..15 / K+16..31
    const int hB   = (lane < 16) ? 0 : 16;   // B-frag: lane halves take K+0..15 / K+16..31

    v8f acc[2][2] = {};                      // [msub][ntile]
    const unsigned short* Arow0 = sA + (l15)      * LDSK;   // M-subtile 0: rows 0..15
    const unsigned short* Arow1 = sA + (16 + l15) * LDSK;   // M-subtile 1: rows 16..31
    const int nbcol = ngrp * 32 + l15;
    const unsigned short* B0 = wlinb + (size_t)(nbcol)      * KTOT + hB;
    const unsigned short* B1 = wlinb + (size_t)(nbcol + 16) * KTOT + hB;

    for (int kk = 0; kk < KTOT; kk += 32) {
        v8bf lo0 = *(const v8bf*)(Arow0 + kk + hA);
        v8bf hi0 = *(const v8bf*)(Arow0 + kk + 8 + hA);
        v16bf a0 = __builtin_shufflevector(lo0, hi0,
                     0, 1, 2, 3, 4, 5, 6, 7, 8, 9, 10, 11, 12, 13, 14, 15);
        v8bf lo1 = *(const v8bf*)(Arow1 + kk + hA);
        v8bf hi1 = *(const v8bf*)(Arow1 + kk + 8 + hA);
        v16bf a1 = __builtin_shufflevector(lo1, hi1,
                     0, 1, 2, 3, 4, 5, 6, 7, 8, 9, 10, 11, 12, 13, 14, 15);
        v16bf b0 = *(const v16bf*)(B0 + kk);
        v16bf b1 = *(const v16bf*)(B1 + kk);
        acc[0][0] = __builtin_amdgcn_wmma_f32_16x16x32_bf16(
                        false, a0, false, b0, (short)0, acc[0][0], false, false);
        acc[0][1] = __builtin_amdgcn_wmma_f32_16x16x32_bf16(
                        false, a0, false, b1, (short)0, acc[0][1], false, false);
        acc[1][0] = __builtin_amdgcn_wmma_f32_16x16x32_bf16(
                        false, a1, false, b0, (short)0, acc[1][0], false, false);
        acc[1][1] = __builtin_amdgcn_wmma_f32_16x16x32_bf16(
                        false, a1, false, b1, (short)0, acc[1][1], false, false);
    }

    // epilogue: + b_lin, swish, streaming NT stores
    #pragma unroll
    for (int ms = 0; ms < 2; ++ms) {
        const int mGlob = eBase + ms * 16 + ((lane < 16) ? 0 : 8);
        #pragma unroll
        for (int nt = 0; nt < 2; ++nt) {
            const int col = ngrp * 32 + nt * 16 + l15;
            const float bl = b_lin[col];
            #pragma unroll
            for (int r = 0; r < 8; ++r) {
                float v = acc[ms][nt][r] + bl;
                v = fast_swish(v);
                __builtin_nontemporal_store(v, &out[(size_t)(mGlob + r) * HID + col]);
            }
        }
    }
}

extern "C" void kernel_launch(void* const* d_in, const int* in_sizes, int n_in,
                              void* d_out, int out_size, void* d_ws, size_t ws_size,
                              hipStream_t stream) {
    const int*   x     = (const int*)  d_in[0];
    const float* cond  = (const float*)d_in[1];
    const float* rbf   = (const float*)d_in[2];
    const int*   ei    = (const int*)  d_in[3];
    const int*   ej    = (const int*)  d_in[4];
    const float* emb_w = (const float*)d_in[5];
    const float* w_rbf = (const float*)d_in[6];
    const float* b_rbf = (const float*)d_in[7];
    const float* w_lin = (const float*)d_in[8];
    const float* b_lin = (const float*)d_in[9];
    float* out = (float*)d_out;

    unsigned short* ws_h    = (unsigned short*)d_ws;                 // 16384*320 bf16 (10.5 MB)
    unsigned short* ws_wlin = ws_h + (size_t)N_NODES * HC;           // 256*896 bf16 (448 KB)

    build_h_kernel<<<N_NODES, HC, 0, stream>>>(x, cond, emb_w, ws_h);
    build_wlin_kernel<<<(HID * KTOT) / 256, 256, 0, stream>>>(w_lin, ws_wlin);
    edge_mlp_kernel<<<N_EDGES / MT, 256, 0, stream>>>(
        ws_h, ws_wlin, ei, ej, rbf, w_rbf, b_rbf, b_lin, out);
}
